// To1D128freq_49331994362237
// MI455X (gfx1250) — matrix operand order repaired and checked by reading, not compiled, not run on hardware
//
#include <hip/hip_runtime.h>
#include <math.h>

typedef __attribute__((ext_vector_type(16))) _Float16 v16h;
typedef __attribute__((ext_vector_type(8)))  float    v8f;
typedef int v4i_ __attribute__((vector_size(16)));   // matches builtin's V4i

#define B_  32
#define CF_ 2048
#define T_  512
#define H_  128
#define H2_ 256
#define H4_ 512

// ---- gfx1250 async global->LDS copies (guarded; fallback = plain copy) ----
#if __has_builtin(__builtin_amdgcn_global_load_async_to_lds_b128) && \
    __has_builtin(__builtin_amdgcn_global_load_async_to_lds_b32)
#define HAVE_ASYNC_LDS 1
#define AS1_ __attribute__((address_space(1)))
#define AS3_ __attribute__((address_space(3)))
#define ASYNC_CP4(g, l)                                         \
    __builtin_amdgcn_global_load_async_to_lds_b32(              \
        (AS1_ int*)(g), (AS3_ int*)(l), 0, 0)
#define ASYNC_CP16(g, l)                                        \
    __builtin_amdgcn_global_load_async_to_lds_b128(             \
        (AS1_ v4i_*)(g), (AS3_ v4i_*)(l), 0, 0)
#endif

__device__ __forceinline__ void wait_async0() {
#if __has_builtin(__builtin_amdgcn_s_wait_asynccnt)
    __builtin_amdgcn_s_wait_asynccnt(0);
#else
    asm volatile("s_wait_asynccnt 0x0" ::: "memory");
#endif
}

__device__ __forceinline__ float sigf(float x) { return 1.0f / (1.0f + __expf(-x)); }

// A-fragment K index for v_wmma_f32_16x16x32_f16 (16-bit A, 16x32):
// lane 0-15: K = {0..7,16..23}; lane 16-31: K = {8..15,24..31}
__device__ __forceinline__ int a_k(int e, int half) {
    return (e & 7) + ((e >> 3) << 4) + (half << 3);
}

// ---------------- prep kernels ----------------
__global__ void k_cvt_f16(const float* __restrict__ src, _Float16* __restrict__ dst, int n) {
    int i = blockIdx.x * blockDim.x + threadIdx.x;
    if (i < n) dst[i] = (_Float16)src[i];
}

__global__ void k_cat_w(const float* __restrict__ Wih, const float* __restrict__ Whh,
                        _Float16* __restrict__ Wcat) {
    int i = blockIdx.x * blockDim.x + threadIdx.x;
    if (i >= H4_ * H2_) return;
    int j = i / H2_, k = i % H2_;
    float v = (k < H_) ? Wih[j * H_ + k] : Whh[j * H_ + (k - H_)];
    Wcat[i] = (_Float16)v;
}

__global__ void k_zero_state(_Float16* hp_f, _Float16* hp_b, float* c_f, float* c_b) {
    int i = blockIdx.x * blockDim.x + threadIdx.x;
    if (i < T_ * H_) {
        hp_f[i] = (_Float16)0.f; hp_b[i] = (_Float16)0.f;
        c_f[i] = 0.f; c_b[i] = 0.f;
    }
}

// ---------------- fc1: hin[b*T+t][d] = sum_c x[b][c][t]*Wfc1[d][c] + bias[d] ----------------
// Block: one 16-column t-tile; 8 waves = 8 d-tiles. x tile staged (async) in LDS, shared by waves.
__global__ void __launch_bounds__(256)
k_fc1(const float* __restrict__ x, const _Float16* __restrict__ Wf,
      const float* __restrict__ bias, _Float16* __restrict__ hin) {
    __shared__ float xsl[16 * 256];   // [n][k] transposed slab, 16 KB

    int tid = threadIdx.x;
    int lane = tid & 31, wid = tid >> 5;
    int half = lane >> 4, mn = lane & 15;
    int nt = blockIdx.x;              // 1024 blocks
    int b  = nt >> 5;
    int t0 = (nt & 31) << 4;
    int m0 = wid << 4;                // d tile

    const float* xb = x + (size_t)b * CF_ * T_ + t0;
    v8f acc = {};
    for (int ks = 0; ks < CF_; ks += 256) {
        // stage 256(K) x 16(t) f32, transposed to [n][k] in LDS
#ifdef HAVE_ASYNC_LDS
#pragma unroll
        for (int r = 0; r < 16; ++r) {
            int linear = r * 256 + tid;
            int k = linear >> 4, n = linear & 15;
            ASYNC_CP4(xb + (size_t)(ks + k) * T_ + n, &xsl[n * 256 + k]);
        }
        wait_async0();
#else
#pragma unroll
        for (int r = 0; r < 16; ++r) {
            int linear = r * 256 + tid;
            int k = linear >> 4, n = linear & 15;
            xsl[n * 256 + k] = xb[(size_t)(ks + k) * T_ + n];
        }
#endif
        __syncthreads();
        if (ks + 256 < CF_)
            __builtin_prefetch(xb + (size_t)(ks + 256 + (tid >> 4)) * T_ + (tid & 15), 0, 1);
#pragma unroll
        for (int kk = 0; kk < 8; ++kk) {
            int k0 = ks + (kk << 5);
            v16h a, bm;
#pragma unroll
            for (int e = 0; e < 16; ++e)
                a[e] = Wf[(size_t)(m0 + mn) * CF_ + k0 + a_k(e, half)];
#pragma unroll
            for (int e = 0; e < 16; ++e)
                bm[e] = (_Float16)xsl[mn * 256 + (kk << 5) + a_k(e, half)];
            acc = __builtin_amdgcn_wmma_f32_16x16x32_f16(false, a, false, bm, (short)0, acc, false, false);
        }
        __syncthreads();
    }
#pragma unroll
    for (int r = 0; r < 8; ++r) {
        int d = m0 + r + (half << 3);
        hin[((size_t)b * T_ + t0 + mn) * H_ + d] = (_Float16)(acc[r] + bias[d]);
    }
}

// ---------------- one LSTM step, both directions ----------------
// z = [x_l | h_prev] (16x256) @ Wcat^T (256x512); A slab async-staged in LDS, gates fused via LDS
__global__ void __launch_bounds__(256)
k_lstm_step(const _Float16* __restrict__ hin,
            const _Float16* __restrict__ Wcat_f, const _Float16* __restrict__ Wcat_b,
            const float* __restrict__ bias_f, const float* __restrict__ bias_b,
            _Float16* __restrict__ hp_f, _Float16* __restrict__ hp_b,
            float* __restrict__ c_f, float* __restrict__ c_b,
            _Float16* __restrict__ hs, int l) {
    __shared__ float    zs[16 * H4_];   // 32 KB gate pre-activations
    __shared__ _Float16 asl[16 * H2_];  // 8 KB  [m][k] = [x_l | h_prev]

    int tid = threadIdx.x;
    int dir = blockIdx.y;
    int t0  = blockIdx.x << 4;
    int lane = tid & 31, wid = tid >> 5;
    int half = lane >> 4, mn = lane & 15;

    int xrow = dir ? (B_ - 1 - l) : l;
    const _Float16* Wcat = dir ? Wcat_b : Wcat_f;
    const float*    bias = dir ? bias_b : bias_f;
    _Float16*       hp   = dir ? hp_b  : hp_f;
    float*          cst  = dir ? c_b   : c_f;

    const _Float16* xa = hin + ((size_t)xrow * T_ + t0) * H_;  // 16 x 128 f16
    const _Float16* ha = hp + (size_t)t0 * H_;                 // 16 x 128 f16

    // stage [x | h] into LDS (16B chunks; dest offset = m*256 + jj*8 for both halves)
#ifdef HAVE_ASYNC_LDS
#pragma unroll
    for (int r = 0; r < 2; ++r) {
        int c = r * 256 + tid;          // 512 chunks of 16B
        int m = c >> 5, jj = c & 31;
        const _Float16* g = (jj < 16) ? (xa + m * H_ + jj * 8)
                                      : (ha + m * H_ + (jj - 16) * 8);
        ASYNC_CP16(g, &asl[m * H2_ + jj * 8]);
    }
    wait_async0();
#else
#pragma unroll
    for (int r = 0; r < 16; ++r) {
        int idx = r * 256 + tid;        // 4096 halves
        int m = idx >> 8, k = idx & 255;
        asl[m * H2_ + k] = (k < H_) ? xa[m * H_ + k] : ha[m * H_ + (k - H_)];
    }
#endif
    __syncthreads();

    for (int i = 0; i < 4; ++i) {       // 8 waves x 4 tiles = 32 N-tiles
        int j0 = ((wid << 2) + i) << 4;
        v8f acc = {};
#pragma unroll
        for (int kk = 0; kk < 8; ++kk) { // K = 256
            int k0 = kk << 5;
            v16h a, bm;
#pragma unroll
            for (int e = 0; e < 16; ++e) a[e] = asl[mn * H2_ + k0 + a_k(e, half)];
#pragma unroll
            for (int e = 0; e < 16; ++e)
                bm[e] = Wcat[(size_t)(j0 + mn) * H2_ + k0 + a_k(e, half)];
            acc = __builtin_amdgcn_wmma_f32_16x16x32_f16(false, a, false, bm, (short)0, acc, false, false);
        }
#pragma unroll
        for (int r = 0; r < 8; ++r)
            zs[(r + (half << 3)) * H4_ + j0 + mn] = acc[r];
    }
    __syncthreads();

    for (int idx = tid; idx < 16 * H_; idx += 256) {
        int tl = idx >> 7, d = idx & 127;
        int t = t0 + tl;
        float zi = zs[tl * H4_ + d]          + bias[d];
        float zf = zs[tl * H4_ + H_ + d]     + bias[H_ + d];
        float zg = zs[tl * H4_ + 2 * H_ + d] + bias[2 * H_ + d];
        float zo = zs[tl * H4_ + 3 * H_ + d] + bias[3 * H_ + d];
        float cp = cst[(size_t)t * H_ + d];
        float cn = sigf(zf) * cp + sigf(zi) * tanhf(zg);
        float hn = sigf(zo) * tanhf(cn);
        cst[(size_t)t * H_ + d] = cn;
        hp[(size_t)t * H_ + d]  = (_Float16)hn;
        hs[((size_t)xrow * T_ + t) * H2_ + dir * H_ + d] = (_Float16)hn;
    }
}

// ---------------- output: out[b][e][t] = sum_d hs[b][t][d]*Wout[e][d] + b_out[e] ----------------
__global__ void __launch_bounds__(256)
k_out(const _Float16* __restrict__ hs, const _Float16* __restrict__ Wo,
      const float* __restrict__ bias, float* __restrict__ out) {
    __shared__ _Float16 asl[16 * H2_];  // 8 KB contiguous hs slab

    int tid = threadIdx.x;
    int lane = tid & 31, wid = tid >> 5;
    int half = lane >> 4, mn = lane & 15;
    int nt = blockIdx.x;               // 1024 blocks
    int b  = nt >> 5;
    int t0 = (nt & 31) << 4;
    int n0 = wid << 4;                 // e tile

    const _Float16* aslab = hs + ((size_t)b * T_ + t0) * H2_;
#ifdef HAVE_ASYNC_LDS
#pragma unroll
    for (int r = 0; r < 2; ++r) {
        int c = r * 256 + tid;
        ASYNC_CP16(aslab + c * 8, &asl[c * 8]);
    }
    wait_async0();
#else
#pragma unroll
    for (int r = 0; r < 16; ++r) {
        int idx = r * 256 + tid;
        asl[idx] = aslab[idx];
    }
#endif
    __syncthreads();

    v8f acc = {};
#pragma unroll
    for (int kk = 0; kk < 8; ++kk) {
        int k0 = kk << 5;
        v16h a, bm;
#pragma unroll
        for (int e = 0; e < 16; ++e)
            a[e] = asl[mn * H2_ + k0 + a_k(e, half)];
#pragma unroll
        for (int e = 0; e < 16; ++e)
            bm[e] = Wo[(size_t)(n0 + mn) * H2_ + k0 + a_k(e, half)];
        acc = __builtin_amdgcn_wmma_f32_16x16x32_f16(false, a, false, bm, (short)0, acc, false, false);
    }
    int eo = n0 + mn;
#pragma unroll
    for (int r = 0; r < 8; ++r) {
        int t = t0 + r + (half << 3);
        out[(size_t)b * H_ * T_ + (size_t)eo * T_ + t] = acc[r] + bias[eo];
    }
}

extern "C" void kernel_launch(void* const* d_in, const int* in_sizes, int n_in,
                              void* d_out, int out_size, void* d_ws, size_t ws_size,
                              hipStream_t stream) {
    (void)in_sizes; (void)n_in; (void)out_size; (void)ws_size;
    const float* x     = (const float*)d_in[0];
    const float* Wfc1  = (const float*)d_in[1];
    const float* bfc1  = (const float*)d_in[2];
    const float* Wih_f = (const float*)d_in[3];
    const float* Whh_f = (const float*)d_in[4];
    const float* b_f   = (const float*)d_in[5];
    const float* Wih_b = (const float*)d_in[6];
    const float* Whh_b = (const float*)d_in[7];
    const float* b_b   = (const float*)d_in[8];
    const float* W_out = (const float*)d_in[9];
    const float* b_out = (const float*)d_in[10];
    float* out = (float*)d_out;

    char* ws = (char*)d_ws;
    size_t off = 0;
    auto alloc = [&](size_t bytes) -> void* {
        void* p = ws + off;
        off += (bytes + 255) & ~(size_t)255;
        return p;
    };
    _Float16* Wf16   = (_Float16*)alloc((size_t)H_ * CF_ * 2);
    _Float16* Wcat_f = (_Float16*)alloc((size_t)H4_ * H2_ * 2);
    _Float16* Wcat_b = (_Float16*)alloc((size_t)H4_ * H2_ * 2);
    _Float16* Wo16   = (_Float16*)alloc((size_t)H_ * H2_ * 2);
    _Float16* hin    = (_Float16*)alloc((size_t)B_ * T_ * H_ * 2);
    _Float16* hs     = (_Float16*)alloc((size_t)B_ * T_ * H2_ * 2);
    _Float16* hp_f   = (_Float16*)alloc((size_t)T_ * H_ * 2);
    _Float16* hp_b   = (_Float16*)alloc((size_t)T_ * H_ * 2);
    float*    c_f    = (float*)alloc((size_t)T_ * H_ * 4);
    float*    c_b    = (float*)alloc((size_t)T_ * H_ * 4);

    k_cvt_f16<<<(H_ * CF_ + 255) / 256, 256, 0, stream>>>(Wfc1, Wf16, H_ * CF_);
    k_cat_w<<<(H4_ * H2_ + 255) / 256, 256, 0, stream>>>(Wih_f, Whh_f, Wcat_f);
    k_cat_w<<<(H4_ * H2_ + 255) / 256, 256, 0, stream>>>(Wih_b, Whh_b, Wcat_b);
    k_cvt_f16<<<(H_ * H2_ + 255) / 256, 256, 0, stream>>>(W_out, Wo16, H_ * H2_);
    k_zero_state<<<(T_ * H_ + 255) / 256, 256, 0, stream>>>(hp_f, hp_b, c_f, c_b);

    k_fc1<<<dim3(B_ * T_ / 16), 256, 0, stream>>>(x, Wf16, bfc1, hin);

    for (int l = 0; l < B_; ++l)
        k_lstm_step<<<dim3(T_ / 16, 2), 256, 0, stream>>>(hin, Wcat_f, Wcat_b, b_f, b_b,
                                                          hp_f, hp_b, c_f, c_b, hs, l);

    k_out<<<dim3(B_ * T_ / 16), 256, 0, stream>>>(hs, Wo16, b_out, out);
}
